// TunnelGrad_45037027066227
// MI455X (gfx1250) — matrix operand order repaired and checked by reading, not compiled, not run on hardware
//
#include <hip/hip_runtime.h>
#include <hip/hip_bf16.h>
#include <math.h>

#define N_SAMPLES   7500
#define NUM_CLASSES 10
#define K_DIM       8000
#define CLIPV       (1.0f - 1e-7f)
#define EPSV        1.1920928955078125e-07f

#define BLK   128
#define KC    64
#define NT    59                           // ceil(7500/128)
#define NROWS_PAD (NT * BLK)               // 7552
#define NPAIR_BLOCKS (NT * (NT + 1) / 2)   // 1770 upper-triangular block pairs
#define NPART 21                           // 1 (S_all) + 10 (W[c]) + 10 (Q[c])
#define LDSTRIDE 72                        // 64 bf16 + 8 pad (144 bytes, 16B aligned)

// ws layout (floats): invnorm[0..7552), mcnt @7552, partials @7568 (1770*21),
// bf16 matrix @ float-offset 44800 (256B aligned) when ws is large enough.
#define WS_XBF_FOFF 44800
#define WS_BIG_BYTES ((size_t)WS_XBF_FOFF * 4 + (size_t)NROWS_PAD * K_DIM * 2)

#if defined(__has_builtin)
#if __has_builtin(__builtin_amdgcn_global_load_async_to_lds_b128)
#define USE_ASYNC 1
#pragma message("CDNA5: global_load_async_to_lds_b128 builtin AVAILABLE - async GEMM path compiled")
#endif
#endif
#ifndef USE_ASYNC
#define USE_ASYNC 0
#pragma message("CDNA5: async builtin NOT available - synchronous GEMM path only")
#endif

typedef __attribute__((ext_vector_type(16))) __bf16 v16bf;
typedef __attribute__((ext_vector_type(8)))  float  v8f;

__device__ __forceinline__ unsigned int pack_bf16x2(float a, float b) {
  unsigned int ua = __float_as_uint(a);
  unsigned int ub = __float_as_uint(b);
  ua += 0x7FFFu + ((ua >> 16) & 1u);   // round-to-nearest-even
  ub += 0x7FFFu + ((ub >> 16) & 1u);
  return (ua >> 16) | (ub & 0xFFFF0000u);
}

// ---------------------------------------------------------------- shared device helpers
// One 64-wide K-chunk of WMMAs: 2x4 tiles, 2 K-steps of 32 -> 16 v_wmma per call.
__device__ __forceinline__ void wmma_chunk(const unsigned short* As_,
                                           const unsigned short* Bs_,
                                           int wr, int wc, int lmod, int lhalf,
                                           v8f (&acc)[2][4]) {
  #pragma unroll
  for (int ks = 0; ks < KC; ks += 32) {
    v16bf fa[2], fb[4];
    // A fragment: lane = row M (mod 16); per lane 16 contiguous K values per half-wave
    #pragma unroll
    for (int a = 0; a < 2; ++a) {
      int m = wr * 32 + a * 16 + lmod;
      union { v16bf v; uint4 q[2]; } u;
      u.q[0] = *(const uint4*)&As_[m * LDSTRIDE + ks + lhalf * 16];
      u.q[1] = *(const uint4*)&As_[m * LDSTRIDE + ks + lhalf * 16 + 8];
      fa[a] = u.v;
    }
    // B fragment: lane = column N; per lane 16 contiguous K values per half-wave
    #pragma unroll
    for (int b = 0; b < 4; ++b) {
      int nn = wc * 64 + b * 16 + lmod;
      union { v16bf v; uint4 q[2]; } u;
      const uint4* p = (const uint4*)&Bs_[nn * LDSTRIDE + ks + lhalf * 16];
      u.q[0] = p[0];
      u.q[1] = p[1];
      fb[b] = u.v;
    }
    #pragma unroll
    for (int a = 0; a < 2; ++a)
      #pragma unroll
      for (int b = 0; b < 4; ++b)
        acc[a][b] = __builtin_amdgcn_wmma_f32_16x16x32_bf16(
            false, fa[a], false, fb[b], (short)0, acc[a][b], false, false);
  }
}

// Fused epilogue: clip+acos on accumulators, masked per-class register bins,
// deterministic fixed-order LDS tree reduction, per-block partial output.
__device__ __forceinline__ void epilogue_reduce(const v8f (&acc)[2][4],
                                                int bi, int bj, int wr, int wc,
                                                int lmod, int lhalf,
                                                const int* yA, const int* yB,
                                                float* red,
                                                float* __restrict__ partials,
                                                int blk_id, int tid) {
  float s_all = 0.f;
  float wsum[NUM_CLASSES], qsum[NUM_CLASSES];
  #pragma unroll
  for (int c = 0; c < NUM_CLASSES; ++c) { wsum[c] = 0.f; qsum[c] = 0.f; }

  #pragma unroll
  for (int a = 0; a < 2; ++a)
    #pragma unroll
    for (int b = 0; b < 4; ++b)
      #pragma unroll
      for (int r8 = 0; r8 < 8; ++r8) {
        int mi = wr * 32 + a * 16 + r8 + 8 * lhalf;   // C layout: M = r + 8*(lane>=16)
        int nj = wc * 64 + b * 16 + lmod;             //           N = lane & 15
        int i = bi * BLK + mi;
        int j = bj * BLK + nj;
        if (i < N_SAMPLES && j < N_SAMPLES && i < j) {
          float cosv = acc[a][b][r8];
          cosv = fminf(fmaxf(cosv, -CLIPV), CLIPV);
          float ang = fabsf(acosf(cosv));
          s_all += ang;
          int yi = yA[mi], yj = yB[nj];
          if (yi == yj) {
            float a2 = ang * ang;
            #pragma unroll
            for (int c = 0; c < NUM_CLASSES; ++c)
              if (yi == c) { wsum[c] += ang; qsum[c] += a2; }
          }
        }
      }

  float vals[NPART];
  vals[0] = s_all;
  #pragma unroll
  for (int c = 0; c < NUM_CLASSES; ++c) { vals[1 + c] = wsum[c]; vals[11 + c] = qsum[c]; }

  #pragma unroll
  for (int comp = 0; comp < NPART; ++comp) {
    __syncthreads();
    red[tid] = vals[comp];
    __syncthreads();
    #pragma unroll
    for (int off = 128; off >= 1; off >>= 1) {
      if (tid < off) red[tid] += red[tid + off];
      __syncthreads();
    }
    if (tid == 0) partials[(size_t)blk_id * NPART + comp] = red[0];
  }
}

// ---------------------------------------------------------------- row norms
__global__ __launch_bounds__(256) void rownorm_kernel(const float* __restrict__ x,
                                                      float* __restrict__ invnorm) {
  const int row = blockIdx.x;
  const float4* xr = (const float4*)(x + (size_t)row * K_DIM);
  float s = 0.f;
  for (int k = threadIdx.x; k < K_DIM / 4; k += 256) {
    float4 v = xr[k];
    s += v.x * v.x + v.y * v.y + v.z * v.z + v.w * v.w;
  }
  __shared__ float red[256];
  red[threadIdx.x] = s;
  __syncthreads();
  #pragma unroll
  for (int off = 128; off >= 1; off >>= 1) {
    if (threadIdx.x < off) red[threadIdx.x] += red[threadIdx.x + off];
    __syncthreads();
  }
  if (threadIdx.x == 0) invnorm[row] = 1.0f / (sqrtf(red[0]) + EPSV);
}

// ---------------------------------------------------------------- class histogram
__global__ void hist_kernel(const int* __restrict__ y, int* __restrict__ mcnt) {
  __shared__ int h[NUM_CLASSES];
  if (threadIdx.x < NUM_CLASSES) h[threadIdx.x] = 0;
  __syncthreads();
  for (int i = threadIdx.x; i < N_SAMPLES; i += blockDim.x) atomicAdd(&h[y[i]], 1);
  __syncthreads();
  if (threadIdx.x < NUM_CLASSES) mcnt[threadIdx.x] = h[threadIdx.x];
}

// ---------------------------------------------------------------- normalize + f32->bf16 (async path)
__global__ __launch_bounds__(256) void tobf16_kernel(const float* __restrict__ x,
                                                     const float* __restrict__ invnorm,
                                                     unsigned short* __restrict__ xbf) {
  const int row = blockIdx.x;                      // 0..NROWS_PAD-1
  unsigned short* dst = xbf + (size_t)row * K_DIM;
  if (row >= N_SAMPLES) {                          // zero padding rows -> branchless GEMM loads
    uint4 z; z.x = z.y = z.z = z.w = 0u;
    uint4* d4 = (uint4*)dst;
    for (int q = threadIdx.x; q < K_DIM / 8; q += 256) d4[q] = z;
    return;
  }
  const float sc = invnorm[row];
  const float4* s4 = (const float4*)(x + (size_t)row * K_DIM);
  uint2* d2 = (uint2*)dst;
  for (int q = threadIdx.x; q < K_DIM / 4; q += 256) {
    float4 v = s4[q];
    uint2 o;
    o.x = pack_bf16x2(v.x * sc, v.y * sc);
    o.y = pack_bf16x2(v.z * sc, v.w * sc);
    d2[q] = o;
  }
}

#if USE_ASYNC
// ---------------------------------------------------------------- async double-buffered GEMM
// Builtin signature (from hipcc diagnostic): pointers to 'int vector_size(16)'
// in global (AS1) / shared (AS3) address spaces.
typedef int v4i_ __attribute__((vector_size(16)));
typedef __attribute__((address_space(1))) v4i_* as1_v4i;
typedef __attribute__((address_space(3))) v4i_* as3_v4i;

__device__ __forceinline__ void async_fill8(as3_v4i dA, const unsigned short* sA,
                                            as3_v4i dB, const unsigned short* sB) {
  as1_v4i a = (as1_v4i)(void*)(const void*)sA;
  as1_v4i b = (as1_v4i)(void*)(const void*)sB;
  __builtin_amdgcn_global_load_async_to_lds_b128(a,     dA,     0, 0);
  __builtin_amdgcn_global_load_async_to_lds_b128(a + 1, dA + 1, 0, 0);
  __builtin_amdgcn_global_load_async_to_lds_b128(a + 2, dA + 2, 0, 0);
  __builtin_amdgcn_global_load_async_to_lds_b128(a + 3, dA + 3, 0, 0);
  __builtin_amdgcn_global_load_async_to_lds_b128(b,     dB,     0, 0);
  __builtin_amdgcn_global_load_async_to_lds_b128(b + 1, dB + 1, 0, 0);
  __builtin_amdgcn_global_load_async_to_lds_b128(b + 2, dB + 2, 0, 0);
  __builtin_amdgcn_global_load_async_to_lds_b128(b + 3, dB + 3, 0, 0);
}

__global__ __launch_bounds__(256) void gram_angle_async_kernel(
    const unsigned short* __restrict__ xbf,   // NROWS_PAD x K_DIM bf16, padded rows zero
    const int* __restrict__ y,
    float* __restrict__ partials) {
  __shared__ __align__(16) unsigned short As[2][BLK * LDSTRIDE];
  __shared__ __align__(16) unsigned short Bs[2][BLK * LDSTRIDE];
  __shared__ int yA[BLK];
  __shared__ int yB[BLK];
  __shared__ float red[256];

  int bi = 0, rem = blockIdx.x;
  while (rem >= NT - bi) { rem -= (NT - bi); ++bi; }
  const int bj = bi + rem;

  const int tid   = threadIdx.x;
  const int wave  = tid >> 5;
  const int lane  = tid & 31;
  const int wr    = wave >> 1;
  const int wc    = wave & 1;
  const int lmod  = lane & 15;
  const int lhalf = lane >> 4;

  if (tid < BLK) {
    int gi = bi * BLK + tid;
    yA[tid] = (gi < N_SAMPLES) ? y[gi] : -1;
  } else {
    int l = tid - BLK;
    int gj = bj * BLK + l;
    yB[l] = (gj < N_SAMPLES) ? y[gj] : -2;
  }

  v8f acc[2][4];
  #pragma unroll
  for (int a = 0; a < 2; ++a)
    #pragma unroll
    for (int b = 0; b < 4; ++b)
      acc[a][b] = v8f{};

  // Each thread async-copies a 64-byte (32 bf16) row segment per panel per chunk
  // as 4x b128 lane transfers. Padding rows are zeros, so no bounds checks.
  // AS3 destination pointers are cast ONCE here so the generic->AS3 null checks
  // stay out of the hot loop; sources advance incrementally (AS1 cast is a no-op).
  const int r    = tid >> 1;
  const int half = tid & 1;
  const unsigned short* srcA = xbf + (size_t)(bi * BLK + r) * K_DIM + half * 32;
  const unsigned short* srcB = xbf + (size_t)(bj * BLK + r) * K_DIM + half * 32;
  as3_v4i dstA3[2] = { (as3_v4i)(void*)&As[0][r * LDSTRIDE + half * 32],
                       (as3_v4i)(void*)&As[1][r * LDSTRIDE + half * 32] };
  as3_v4i dstB3[2] = { (as3_v4i)(void*)&Bs[0][r * LDSTRIDE + half * 32],
                       (as3_v4i)(void*)&Bs[1][r * LDSTRIDE + half * 32] };

  // prologue: fill buffer 0 with chunk k0 = 0
  async_fill8(dstA3[0], srcA, dstB3[0], srcB);
  asm volatile("s_wait_asynccnt 0x0" ::: "memory");
  __syncthreads();

  int cur = 0;
  for (int k0 = 0; k0 < K_DIM; k0 += KC) {
    const int nxt = cur ^ 1;
    if (k0 + KC < K_DIM) {
      srcA += KC;
      srcB += KC;
      async_fill8(dstA3[nxt], srcA, dstB3[nxt], srcB);
    }
    // WMMAs on current buffer overlap the in-flight async fills
    wmma_chunk(&As[cur][0], &Bs[cur][0], wr, wc, lmod, lhalf, acc);
    asm volatile("s_wait_asynccnt 0x0" ::: "memory");
    __syncthreads();
    cur = nxt;
  }

  epilogue_reduce(acc, bi, bj, wr, wc, lmod, lhalf, yA, yB, red, partials, blockIdx.x, tid);
}
#endif  // USE_ASYNC

// ---------------------------------------------------------------- synchronous fallback GEMM
__global__ __launch_bounds__(256) void gram_angle_kernel(const float* __restrict__ x,
                                                         const int* __restrict__ y,
                                                         const float* __restrict__ invnorm,
                                                         float* __restrict__ partials) {
  __shared__ __align__(16) unsigned short As[BLK * LDSTRIDE];
  __shared__ __align__(16) unsigned short Bs[BLK * LDSTRIDE];
  __shared__ int yA[BLK];
  __shared__ int yB[BLK];
  __shared__ float red[256];

  int bi = 0, rem = blockIdx.x;
  while (rem >= NT - bi) { rem -= (NT - bi); ++bi; }
  const int bj = bi + rem;

  const int tid   = threadIdx.x;
  const int wave  = tid >> 5;
  const int lane  = tid & 31;
  const int wr    = wave >> 1;
  const int wc    = wave & 1;
  const int lmod  = lane & 15;
  const int lhalf = lane >> 4;

  if (tid < BLK) {
    int gi = bi * BLK + tid;
    yA[tid] = (gi < N_SAMPLES) ? y[gi] : -1;
  } else {
    int l = tid - BLK;
    int gj = bj * BLK + l;
    yB[l] = (gj < N_SAMPLES) ? y[gj] : -2;
  }

  v8f acc[2][4];
  #pragma unroll
  for (int a = 0; a < 2; ++a)
    #pragma unroll
    for (int b = 0; b < 4; ++b)
      acc[a][b] = v8f{};

  const int r  = tid >> 1;
  const int c0 = (tid & 1) * 32;
  for (int k0 = 0; k0 < K_DIM; k0 += KC) {
    __syncthreads();
    {
      int gi = bi * BLK + r;
      uint4* dst = (uint4*)&As[r * LDSTRIDE + c0];
      if (gi < N_SAMPLES) {
        const float4* s4 = (const float4*)(x + (size_t)gi * K_DIM + k0 + c0);
        if (k0 + KC < K_DIM) __builtin_prefetch((const float*)s4 + KC, 0, 1);
        float sc = invnorm[gi];
        #pragma unroll
        for (int q = 0; q < 4; ++q) {
          float4 v0 = s4[2 * q], v1 = s4[2 * q + 1];
          uint4 o;
          o.x = pack_bf16x2(v0.x * sc, v0.y * sc);
          o.y = pack_bf16x2(v0.z * sc, v0.w * sc);
          o.z = pack_bf16x2(v1.x * sc, v1.y * sc);
          o.w = pack_bf16x2(v1.z * sc, v1.w * sc);
          dst[q] = o;
        }
      } else {
        uint4 z; z.x = z.y = z.z = z.w = 0u;
        #pragma unroll
        for (int q = 0; q < 4; ++q) dst[q] = z;
      }
    }
    {
      int gj = bj * BLK + r;
      uint4* dst = (uint4*)&Bs[r * LDSTRIDE + c0];
      if (gj < N_SAMPLES) {
        const float4* s4 = (const float4*)(x + (size_t)gj * K_DIM + k0 + c0);
        if (k0 + KC < K_DIM) __builtin_prefetch((const float*)s4 + KC, 0, 1);
        float sc = invnorm[gj];
        #pragma unroll
        for (int q = 0; q < 4; ++q) {
          float4 v0 = s4[2 * q], v1 = s4[2 * q + 1];
          uint4 o;
          o.x = pack_bf16x2(v0.x * sc, v0.y * sc);
          o.y = pack_bf16x2(v0.z * sc, v0.w * sc);
          o.z = pack_bf16x2(v1.x * sc, v1.y * sc);
          o.w = pack_bf16x2(v1.z * sc, v1.w * sc);
          dst[q] = o;
        }
      } else {
        uint4 z; z.x = z.y = z.z = z.w = 0u;
        #pragma unroll
        for (int q = 0; q < 4; ++q) dst[q] = z;
      }
    }
    __syncthreads();
    wmma_chunk(As, Bs, wr, wc, lmod, lhalf, acc);
  }

  epilogue_reduce(acc, bi, bj, wr, wc, lmod, lhalf, yA, yB, red, partials, blockIdx.x, tid);
}

// ---------------------------------------------------------------- finalize
__global__ void finalize_kernel(const float* __restrict__ partials,
                                const int* __restrict__ mcnt,
                                float* __restrict__ out) {
  __shared__ double comp[NPART];
  int tid = threadIdx.x;
  if (tid < NPART) {
    double s = 0.0;
    for (int b = 0; b < NPAIR_BLOCKS; ++b) s += (double)partials[(size_t)b * NPART + tid];
    comp[tid] = s;
  }
  __syncthreads();
  if (tid == 0) {
    double S_all = comp[0];
    double Wtot = 0.0;
    for (int c = 0; c < NUM_CLASSES; ++c) Wtot += comp[1 + c];
    const double n = (double)N_SAMPLES;
    const double n_pairs = n * (n - 1.0) / 2.0;
    // general_sum = same-class ordered pairs (2*Wtot) + all i<j pairs (S_all)
    double general_mean = (S_all + 2.0 * Wtot) / n_pairs;
    double within_cov = 0.0, between_cov = 0.0;
    for (int c = 0; c < NUM_CLASSES; ++c) {
      double m = (double)mcnt[c];
      double cntc = m * (m - 1.0) / 2.0;
      double mean_c = comp[1 + c] / cntc;
      double var_c = comp[11 + c] / cntc - mean_c * mean_c;
      within_cov += var_c;
      double d = general_mean - mean_c;
      between_cov += d * d;
    }
    within_cov /= NUM_CLASSES;
    between_cov /= NUM_CLASSES;
    double total = within_cov + between_cov;
    out[0] = (float)(within_cov / total);
    out[1] = (float)(between_cov / total);
  }
}

// ---------------------------------------------------------------- launch
extern "C" void kernel_launch(void* const* d_in, const int* in_sizes, int n_in,
                              void* d_out, int out_size, void* d_ws, size_t ws_size,
                              hipStream_t stream) {
  const float* x = (const float*)d_in[0];
  const int*   y = (const int*)d_in[1];
  float* ws       = (float*)d_ws;
  float* invnorm  = ws;                       // 7552 floats (7500 used)
  int*   mcnt     = (int*)(ws + 7552);        // 10 ints
  float* partials = ws + 7568;                // 1770 * 21 floats

  rownorm_kernel<<<N_SAMPLES, 256, 0, stream>>>(x, invnorm);
  hist_kernel<<<1, 256, 0, stream>>>(y, mcnt);

#if USE_ASYNC
  if (ws_size >= WS_BIG_BYTES) {
    unsigned short* xbf = (unsigned short*)(ws + WS_XBF_FOFF);
    tobf16_kernel<<<NROWS_PAD, 256, 0, stream>>>(x, invnorm, xbf);
    gram_angle_async_kernel<<<NPAIR_BLOCKS, 256, 0, stream>>>(xbf, y, partials);
  } else {
    gram_angle_kernel<<<NPAIR_BLOCKS, 256, 0, stream>>>(x, y, invnorm, partials);
  }
#else
  gram_angle_kernel<<<NPAIR_BLOCKS, 256, 0, stream>>>(x, y, invnorm, partials);
#endif

  finalize_kernel<<<1, 32, 0, stream>>>(partials, mcnt, (float*)d_out);
}
